// CondOTGRNModel_26036091748503
// MI455X (gfx1250) — matrix operand order repaired and checked by reading, not compiled, not run on hardware
//
#include <hip/hip_runtime.h>
#include <hip/hip_bf16.h>
#include <math.h>

// ---------------------------------------------------------------------------
// Model dims
// ---------------------------------------------------------------------------
#define B_   1024
#define G_   5000
#define P_   1024
#define L_   128
#define CH_  64
#define H_   512
#define EPS_ 0.1f
#define ITERS_ 50

typedef __attribute__((ext_vector_type(16))) __bf16 v16bf;
typedef __attribute__((ext_vector_type(8)))  float  v8f;

__device__ __forceinline__ float gelu_f(float x) {
    // exact erf GELU (torch nn.GELU default)
    return 0.5f * x * (1.0f + erff(x * 0.70710678118654752f));
}

// ---------------------------------------------------------------------------
// Generic GEMM: C[M,N] = epilogue(A[M,K] @ W[K,N] + bias)
//   A fp32 (lda), W fp32 row-major [K,N], bf16 WMMA compute, f32 accumulate.
//   epilogue: v = acc + bias; act(v); v *= out_scale; if(resid) v += resid.
//   act: 0=none 1=gelu 2=tanh
// Block = 256 threads = 8 wave32s; tile 64x64, BK=64 (4 WMMA per stage).
// Wave w: rows (w&3)*16, cols (w>>2)*32 -> two 16x16 f32 accumulators.
// Staging fast path: float4 global loads -> packed bf16 -> uint4 (b128) DS
// stores. Edge fallback is branchless: clamped addresses + select-zero.
// ---------------------------------------------------------------------------
__global__ __launch_bounds__(256)
void gemm_bf16_wmma(const float* __restrict__ A, int lda,
                    const float* __restrict__ W,
                    const float* __restrict__ bias,
                    float* __restrict__ C, int ldc,
                    const float* __restrict__ resid, int ldr,
                    float out_scale, int act,
                    int M, int N, int K)
{
    __shared__ alignas(16) __bf16 As[64][80];   // 64 x 64 (+16 pad, 16B-aligned rows)
    __shared__ alignas(16) __bf16 Bs[64][80];

    const int tid  = threadIdx.x;
    const int lane = tid & 31;
    const int wave = tid >> 5;
    const int wm   = wave & 3;
    const int wn   = wave >> 2;
    const int bm   = blockIdx.y * 64;
    const int bn   = blockIdx.x * 64;

    v8f acc0 = {};
    v8f acc1 = {};

    // staging maps: each thread stages 16 contiguous elements of one row
    const int sr = tid >> 2;          // 0..63 (row of tile)
    const int sc = (tid & 3) << 4;    // 0,16,32,48 (col group within K/N tile)

    const bool aRowOk  = (bm + sr) < M;
    const int  arowClamp = aRowOk ? (bm + sr) : 0;       // clamped A row
    const bool bColsOk = (bn + 64) <= N;                 // full N tile for B fast path

    for (int k0 = 0; k0 < K; k0 += 64) {
        const bool kFull = (k0 + 64) <= K;

        // ---- stage A tile ----
        if (kFull && aRowOk) {
            const float* src = A + (size_t)(bm + sr) * lda + k0 + sc;
            __builtin_prefetch(src + 64, 0, 1);   // next K tile -> global_prefetch_b8
            float4 f0 = ((const float4*)src)[0];
            float4 f1 = ((const float4*)src)[1];
            float4 f2 = ((const float4*)src)[2];
            float4 f3 = ((const float4*)src)[3];
            union { uint4 u; __bf16 h[8]; } p0, p1;
            p0.h[0]=(__bf16)f0.x; p0.h[1]=(__bf16)f0.y; p0.h[2]=(__bf16)f0.z; p0.h[3]=(__bf16)f0.w;
            p0.h[4]=(__bf16)f1.x; p0.h[5]=(__bf16)f1.y; p0.h[6]=(__bf16)f1.z; p0.h[7]=(__bf16)f1.w;
            p1.h[0]=(__bf16)f2.x; p1.h[1]=(__bf16)f2.y; p1.h[2]=(__bf16)f2.z; p1.h[3]=(__bf16)f2.w;
            p1.h[4]=(__bf16)f3.x; p1.h[5]=(__bf16)f3.y; p1.h[6]=(__bf16)f3.z; p1.h[7]=(__bf16)f3.w;
            *(uint4*)&As[sr][sc]     = p0.u;
            *(uint4*)&As[sr][sc + 8] = p1.u;
        } else {
            // branchless edge path: clamped address, select-zero, no unroll
            const float* arow = A + (size_t)arowClamp * lda;
            for (int j = 0; j < 16; ++j) {
                int gk = k0 + sc + j;
                bool ok = aRowOk && (gk < K);
                float v = arow[ok ? gk : 0];
                As[sr][sc + j] = (__bf16)(ok ? v : 0.0f);
            }
        }

        // ---- stage B tile ----
        if (kFull && bColsOk) {
            const float* src = W + (size_t)(k0 + sr) * N + bn + sc;
            __builtin_prefetch(src + (size_t)64 * N, 0, 1);
            float4 f0 = ((const float4*)src)[0];
            float4 f1 = ((const float4*)src)[1];
            float4 f2 = ((const float4*)src)[2];
            float4 f3 = ((const float4*)src)[3];
            union { uint4 u; __bf16 h[8]; } p0, p1;
            p0.h[0]=(__bf16)f0.x; p0.h[1]=(__bf16)f0.y; p0.h[2]=(__bf16)f0.z; p0.h[3]=(__bf16)f0.w;
            p0.h[4]=(__bf16)f1.x; p0.h[5]=(__bf16)f1.y; p0.h[6]=(__bf16)f1.z; p0.h[7]=(__bf16)f1.w;
            p1.h[0]=(__bf16)f2.x; p1.h[1]=(__bf16)f2.y; p1.h[2]=(__bf16)f2.z; p1.h[3]=(__bf16)f2.w;
            p1.h[4]=(__bf16)f3.x; p1.h[5]=(__bf16)f3.y; p1.h[6]=(__bf16)f3.z; p1.h[7]=(__bf16)f3.w;
            *(uint4*)&Bs[sr][sc]     = p0.u;
            *(uint4*)&Bs[sr][sc + 8] = p1.u;
        } else {
            int gk = k0 + sr;
            bool rowOk = gk < K;
            const float* wrow = W + (size_t)(rowOk ? gk : 0) * N;
            for (int j = 0; j < 16; ++j) {
                int gc = bn + sc + j;
                bool ok = rowOk && (gc < N);
                float v = wrow[ok ? gc : 0];
                Bs[sr][sc + j] = (__bf16)(ok ? v : 0.0f);
            }
        }
        __syncthreads();

        // ---- 2 k-steps x 2 n-halves = 4 WMMA per stage ----
        const int m  = wm * 16 + (lane & 15);
        const int kb = (lane >> 4) * 8;
        const int kr = (lane & 15) + ((lane >> 4) * 16);
        const int cb = wn * 32;
        #pragma unroll
        for (int ks = 0; ks < 2; ++ks) {
            const int kof = ks * 32;
            v16bf af, b0, b1;
            #pragma unroll
            for (int i = 0; i < 8; ++i) af[i]     = As[m][kof + kb + i];
            #pragma unroll
            for (int i = 0; i < 8; ++i) af[8 + i] = As[m][kof + 16 + kb + i];
            #pragma unroll
            for (int i = 0; i < 16; ++i) b0[i] = Bs[kof + (i < 8 ? kb + i : 16 + kb + i - 8)][cb + (lane & 15)];
            #pragma unroll
            for (int i = 0; i < 16; ++i) b1[i] = Bs[kof + (i < 8 ? kb + i : 16 + kb + i - 8)][cb + 16 + (lane & 15)];
            acc0 = __builtin_amdgcn_wmma_f32_16x16x32_bf16(false, af, false, b0,
                                                           (short)0, acc0, false, false);
            acc1 = __builtin_amdgcn_wmma_f32_16x16x32_bf16(false, af, false, b1,
                                                           (short)0, acc1, false, false);
        }
        __syncthreads();
    }

    // epilogue; C/D layout: VGPR r -> row = r + 8*(lane>>4), col = lane&15
    const int rowbase = bm + wm * 16 + ((lane >> 4) * 8);
    const int col0    = bn + wn * 32 + (lane & 15);
    #pragma unroll
    for (int r = 0; r < 8; ++r) {
        int gr = rowbase + r;
        if (gr >= M) continue;
        if (col0 < N) {
            float v = acc0[r] + (bias ? bias[col0] : 0.0f);
            if (act == 1) v = gelu_f(v); else if (act == 2) v = tanhf(v);
            v *= out_scale;
            if (resid) v += resid[(size_t)gr * ldr + col0];
            C[(size_t)gr * ldc + col0] = v;
        }
        int c1 = col0 + 16;
        if (c1 < N) {
            float v = acc1[r] + (bias ? bias[c1] : 0.0f);
            if (act == 1) v = gelu_f(v); else if (act == 2) v = tanhf(v);
            v *= out_scale;
            if (resid) v += resid[(size_t)gr * ldr + c1];
            C[(size_t)gr * ldc + c1] = v;
        }
    }
}

// ---------------------------------------------------------------------------
// Row LayerNorm (+optional exact GELU). One wave32 per row, 8 rows/block.
// ---------------------------------------------------------------------------
__global__ __launch_bounds__(256)
void layernorm_rows(const float* __restrict__ X, int ldx,
                    const float* __restrict__ g, const float* __restrict__ b,
                    float* __restrict__ Y, int ldy, int M, int D, int act)
{
    const int wave = threadIdx.x >> 5;
    const int lane = threadIdx.x & 31;
    const int row  = blockIdx.x * 8 + wave;
    if (row >= M) return;
    const float* x = X + (size_t)row * ldx;
    float s = 0.0f, ss = 0.0f;
    for (int j = lane; j < D; j += 32) { float v = x[j]; s += v; ss += v * v; }
    #pragma unroll
    for (int o = 16; o > 0; o >>= 1) {
        s  += __shfl_xor(s, o, 32);
        ss += __shfl_xor(ss, o, 32);
    }
    const float mean = s / (float)D;
    const float var  = ss / (float)D - mean * mean;
    const float rstd = rsqrtf(var + 1e-5f);
    float* y = Y + (size_t)row * ldy;
    for (int j = lane; j < D; j += 32) {
        float v = g[j] * (x[j] - mean) * rstd + b[j];
        if (act) v = gelu_f(v);
        y[j] = v;
    }
}

// ---------------------------------------------------------------------------
// Dose MLP (1 -> 32 -> 32), output into fused buffer cols [256,288), ld=288
// ---------------------------------------------------------------------------
__global__ void dose_d1(const float* __restrict__ dose,
                        const float* __restrict__ w, const float* __restrict__ b,
                        float* __restrict__ out)
{
    int i = blockIdx.x * blockDim.x + threadIdx.x;
    if (i >= B_ * 32) return;
    int bi = i >> 5, j = i & 31;
    out[i] = gelu_f(dose[bi] * w[j] + b[j]);
}

__global__ void dose_d2_to_fused(const float* __restrict__ in,
                                 const float* __restrict__ w, const float* __restrict__ b,
                                 float* __restrict__ fused)
{
    int i = blockIdx.x * blockDim.x + threadIdx.x;
    if (i >= B_ * 32) return;
    int bi = i >> 5, j = i & 31;
    float s = b[j];
    #pragma unroll
    for (int k = 0; k < 32; ++k) s += in[bi * 32 + k] * w[k * 32 + j];
    fused[(size_t)bi * 288 + 256 + j] = gelu_f(s);
}

// ---------------------------------------------------------------------------
// Strided 2D copy (concat assembly)
// ---------------------------------------------------------------------------
__global__ void copy2d(const float* __restrict__ src, int lds_,
                       float* __restrict__ dst, int ldd, int M, int ncols)
{
    int i = blockIdx.x * blockDim.x + threadIdx.x;
    if (i >= M * ncols) return;
    int r = i / ncols, c = i - r * ncols;
    dst[(size_t)r * ldd + c] = src[(size_t)r * lds_ + c];
}

__global__ void zero1d(float* __restrict__ p, int n)
{
    int i = blockIdx.x * blockDim.x + threadIdx.x;
    if (i < n) p[i] = 0.0f;
}

// ---------------------------------------------------------------------------
// Sinkhorn with row-constant cost: C[i,j] = c[i].
//   u_i' = EPS*loga - (u_i - c_i) - EPS*LSE_j(v_j/EPS)
//   v_j' = EPS*loga - v_j       - EPS*LSE_i((u_i'-c_i)/EPS)
// Single block, 1024 threads, 50 iterations fully in LDS.
// ---------------------------------------------------------------------------
__global__ __launch_bounds__(1024)
void sinkhorn_kernel(const float* __restrict__ cvec,
                     float* __restrict__ u_out, float* __restrict__ v_out)
{
    __shared__ float red[1024];
    const int t = threadIdx.x;
    const float c = cvec[t];
    const float loga = logf(1.0f / 1024.0f + 1e-8f);
    float u = 0.0f, v = 0.0f;

    for (int it = 0; it < ITERS_; ++it) {
        // LSE over v/EPS
        float x = v / EPS_;
        red[t] = x; __syncthreads();
        for (int o = 512; o > 0; o >>= 1) {
            if (t < o) red[t] = fmaxf(red[t], red[t + o]);
            __syncthreads();
        }
        float mx = red[0]; __syncthreads();
        red[t] = expf(x - mx); __syncthreads();
        for (int o = 512; o > 0; o >>= 1) {
            if (t < o) red[t] += red[t + o];
            __syncthreads();
        }
        float lse = logf(red[0]) + mx; __syncthreads();
        u = EPS_ * loga - (u - c) - EPS_ * lse;

        // LSE over (u-c)/EPS
        x = (u - c) / EPS_;
        red[t] = x; __syncthreads();
        for (int o = 512; o > 0; o >>= 1) {
            if (t < o) red[t] = fmaxf(red[t], red[t + o]);
            __syncthreads();
        }
        mx = red[0]; __syncthreads();
        red[t] = expf(x - mx); __syncthreads();
        for (int o = 512; o > 0; o >>= 1) {
            if (t < o) red[t] += red[t + o];
            __syncthreads();
        }
        lse = logf(red[0]) + mx; __syncthreads();
        v = EPS_ * loga - v - EPS_ * lse;
    }
    u_out[t] = u;
    v_out[t] = v;
}

__global__ void pplan_kernel(const float* __restrict__ u, const float* __restrict__ v,
                             const float* __restrict__ cvec, float* __restrict__ P)
{
    int i = blockIdx.x * blockDim.x + threadIdx.x;
    if (i >= B_ * B_) return;
    int r = i >> 10, cI = i & 1023;
    P[i] = expf((u[r] + v[cI] - cvec[r]) / EPS_);
}

// ---------------------------------------------------------------------------
// Flow coupling layers. mask[j] = 1 if (j%2 == parity) (kept positions).
// prepare: si = [ z[:, :64]*(1-mask), p ];  z2 = z[:, 64:]*(1-mask)
// finish:  z[:,64+j] = z[:,64+j]*m + (z2*exp(scale)+trans)*(1-m); ldj += sum(scale)
// ---------------------------------------------------------------------------
__global__ void flow_prepare(const float* __restrict__ z, const float* __restrict__ p,
                             float* __restrict__ si, float* __restrict__ z2, int parity)
{
    int i = blockIdx.x * blockDim.x + threadIdx.x;
    if (i >= B_ * 64) return;
    int bi = i >> 6, j = i & 63;
    float unm = ((j & 1) != parity) ? 1.0f : 0.0f;
    si[(size_t)bi * 128 + j]      = z[(size_t)bi * 128 + j] * unm;
    si[(size_t)bi * 128 + 64 + j] = p[(size_t)bi * 64 + j];
    z2[i] = z[(size_t)bi * 128 + 64 + j] * unm;
}

__global__ __launch_bounds__(256)
void flow_finish(float* __restrict__ z, const float* __restrict__ z2,
                 const float* __restrict__ scale, const float* __restrict__ trans,
                 float* __restrict__ ldj, int parity)
{
    const int wave = threadIdx.x >> 5;
    const int lane = threadIdx.x & 31;
    const int row  = blockIdx.x * 8 + wave;
    if (row >= B_) return;
    float ssum = 0.0f;
    for (int j = lane; j < 64; j += 32) {
        float sc = scale[(size_t)row * 64 + j];
        ssum += sc;
        float m   = ((j & 1) == parity) ? 1.0f : 0.0f;
        float z2n = z2[(size_t)row * 64 + j] * expf(sc) + trans[(size_t)row * 64 + j];
        float old = z[(size_t)row * 128 + 64 + j];
        z[(size_t)row * 128 + 64 + j] = old * m + z2n * (1.0f - m);
    }
    #pragma unroll
    for (int o = 16; o > 0; o >>= 1) ssum += __shfl_xor(ssum, o, 32);
    if (lane == 0) ldj[row] += ssum;
}

// ---------------------------------------------------------------------------
// Host orchestration
// ---------------------------------------------------------------------------
static inline void gemm(hipStream_t s, const float* A, int lda, const float* W,
                        const float* bias, float* C, int ldc,
                        const float* resid, int ldr, float oscale, int act,
                        int M, int N, int K)
{
    dim3 grid((N + 63) / 64, (M + 63) / 64);
    gemm_bf16_wmma<<<grid, 256, 0, s>>>(A, lda, W, bias, C, ldc, resid, ldr,
                                        oscale, act, M, N, K);
}

static inline void ln(hipStream_t s, const float* X, int ldx, const float* g,
                      const float* b, float* Y, int ldy, int M, int D, int act)
{
    layernorm_rows<<<(M + 7) / 8, 256, 0, s>>>(X, ldx, g, b, Y, ldy, M, D, act);
}

extern "C" void kernel_launch(void* const* d_in, const int* in_sizes, int n_in,
                              void* d_out, int out_size, void* d_ws, size_t ws_size,
                              hipStream_t stream)
{
    (void)in_sizes; (void)out_size; (void)ws_size;
    if (n_in < 174) return; // pytree flatten mismatch guard

    auto in = [&](int i) -> const float* { return (const float*)d_in[i]; };

    const float* x_control = in(0);
    const float* pert      = in(1);
    const float* dose      = in(2);
    const float* x_pert    = in(3);
    // params: jax tree_leaves order (dicts sorted alphabetically)
    // chem: 4..19 | dec: 20..89 | enc: 90..121 | flow: 122..169 | ot: 170..173

    // workspace layout (floats)
    float* ws = (float*)d_ws;
    size_t off = 0;
    auto alloc = [&](size_t n) { float* p = ws + off; off += n; return p; };
    float* h     = alloc((size_t)B_ * H_);
    float* x1    = alloc((size_t)B_ * H_);
    float* x2    = alloc((size_t)B_ * H_);
    float* big   = alloc((size_t)B_ * 2048);
    float* zp    = alloc((size_t)B_ * L_);
    float* zb    = alloc((size_t)B_ * L_);
    float* ci    = alloc((size_t)B_ * 192);
    float* fused = alloc((size_t)B_ * 288);
    float* df1b  = alloc((size_t)B_ * 32);
    float* cvec  = alloc((size_t)B_);
    float* ubuf  = alloc((size_t)B_);
    float* vbuf  = alloc((size_t)B_);
    float* si    = alloc((size_t)B_ * 128);
    float* z2b   = alloc((size_t)B_ * 64);
    float* scb   = alloc((size_t)B_ * 64);
    float* trb   = alloc((size_t)B_ * 64);

    // output layout (flat, return order)
    float* out    = (float*)d_out;
    float* o_xp   = out;
    float* o_zc   = o_xp + (size_t)B_ * G_;
    float* o_z    = o_zc + (size_t)B_ * L_;
    float* o_p    = o_z  + (size_t)B_ * L_;
    float* o_P    = o_p  + (size_t)B_ * CH_;
    float* o_ldj  = o_P  + (size_t)B_ * B_;

    // ---------------- gene encoder (run twice) ----------------
    auto encode = [&](const float* x, float* zout) {
        gemm(stream, x, G_, in(91), in(90), h, H_, nullptr, 0, 1.0f, 0, B_, H_, G_);
        for (int l = 0; l < 2; ++l) {
            const int p0 = 92 + 12 * l; // f1.b f1.w f2.b f2.w ln1.b ln1.g ln2.b ln2.g wo.b wo.w wv.b wv.w
            ln(stream, h, H_, in(p0 + 5), in(p0 + 4), x1, H_, B_, H_, 0);           // ln1
            gemm(stream, x1, H_, in(p0 + 11), in(p0 + 10), x2, H_, nullptr, 0, 1.0f, 0, B_, H_, H_); // wv
            gemm(stream, x2, H_, in(p0 + 9),  in(p0 + 8),  h, H_, h, H_, 1.0f, 0, B_, H_, H_);       // h += wo(.)
            ln(stream, h, H_, in(p0 + 7), in(p0 + 6), x1, H_, B_, H_, 0);           // ln2
            gemm(stream, x1, H_, in(p0 + 1), in(p0 + 0), big, 2048, nullptr, 0, 1.0f, 1, B_, 2048, H_); // gelu(f1)
            gemm(stream, big, 2048, in(p0 + 3), in(p0 + 2), h, H_, h, H_, 1.0f, 0, B_, H_, 2048);       // h += f2(.)
        }
        gemm(stream, h, H_, in(117), in(116), x2, 256, nullptr, 0, 1.0f, 0, B_, 256, H_); // out1
        ln(stream, x2, 256, in(121), in(120), x1, 256, B_, 256, 1);                       // gelu(out_ln)
        gemm(stream, x1, 256, in(119), in(118), zout, L_, nullptr, 0, 1.0f, 0, B_, L_, 256); // out2
    };

    encode(x_control, o_zc);  // z_c -> output
    encode(x_pert,    zp);    // z_p -> workspace

    // ---------------- chem branch ----------------
    gemm(stream, pert, P_, in(13), in(12), x1, H_, nullptr, 0, 1.0f, 0, B_, H_, P_); // p1
    ln(stream, x1, H_, in(15), in(14), x2, H_, B_, H_, 1);                            // gelu(p1_ln)
    gemm(stream, x2, H_, in(17), in(16), x1, 256, nullptr, 0, 1.0f, 0, B_, 256, H_);  // p2
    ln(stream, x1, 256, in(19), in(18), fused, 288, B_, 256, 1);                      // pf -> fused[:, :256]
    dose_d1<<<(B_ * 32 + 255) / 256, 256, 0, stream>>>(dose, in(5), in(4), df1b);
    dose_d2_to_fused<<<(B_ * 32 + 255) / 256, 256, 0, stream>>>(df1b, in(7), in(6), fused);
    gemm(stream, fused, 288, in(9), in(8), x1, CH_, nullptr, 0, 1.0f, 0, B_, CH_, 288); // fus
    ln(stream, x1, CH_, in(11), in(10), o_p, CH_, B_, CH_, 1);                          // p -> output

    // ---------------- OT cost + Sinkhorn ----------------
    copy2d<<<(B_ * L_ + 255) / 256, 256, 0, stream>>>(o_zc, L_, ci, 192, B_, L_);
    copy2d<<<(B_ * CH_ + 255) / 256, 256, 0, stream>>>(o_p, CH_, ci + L_, 192, B_, CH_);
    gemm(stream, ci, 192, in(171), in(170), x1, L_, nullptr, 0, 1.0f, 1, B_, L_, 192); // gelu(c1)
    gemm(stream, x1, L_, in(173), in(172), cvec, 1, nullptr, 0, 1.0f, 0, B_, 1, L_);   // c2 -> c[i]
    sinkhorn_kernel<<<1, 1024, 0, stream>>>(cvec, ubuf, vbuf);
    pplan_kernel<<<(B_ * B_ + 255) / 256, 256, 0, stream>>>(ubuf, vbuf, cvec, o_P);
    // z = Pplan @ z_p  (z_p is exactly the [K=1024, N=128] row-major operand)
    gemm(stream, o_P, B_, zp, nullptr, zb, L_, nullptr, 0, 1.0f, 0, B_, L_, B_);

    // ---------------- normalizing flow ----------------
    zero1d<<<(B_ + 255) / 256, 256, 0, stream>>>(o_ldj, B_);
    for (int i = 0; i < 6; ++i) {
        const int p0  = 122 + 8 * i; // s1.b s1.w s2.b s2.w t1.b t1.w t2.b t2.w
        const int par = i & 1;
        flow_prepare<<<(B_ * 64 + 255) / 256, 256, 0, stream>>>(zb, o_p, si, z2b, par);
        gemm(stream, si, 128, in(p0 + 1), in(p0 + 0), x1, H_, nullptr, 0, 1.0f, 1, B_, H_, 128); // gelu(s1)
        gemm(stream, x1, H_, in(p0 + 3), in(p0 + 2), scb, 64, nullptr, 0, 1.0f, 2, B_, 64, H_);  // tanh(s2)
        gemm(stream, si, 128, in(p0 + 5), in(p0 + 4), x1, H_, nullptr, 0, 1.0f, 1, B_, H_, 128); // gelu(t1)
        gemm(stream, x1, H_, in(p0 + 7), in(p0 + 6), trb, 64, nullptr, 0, 1.0f, 0, B_, 64, H_);  // t2
        flow_finish<<<(B_ + 7) / 8, 256, 0, stream>>>(zb, z2b, scb, trb, o_ldj, par);
    }
    copy2d<<<(B_ * L_ + 255) / 256, 256, 0, stream>>>(zb, L_, o_z, L_, B_, L_); // z -> output

    // ---------------- decoder ----------------
    gemm(stream, zb, L_, in(83), in(82), x1, H_, nullptr, 0, 1.0f, 0, B_, H_, L_);  // m1
    ln(stream, x1, H_, in(85), in(84), x2, H_, B_, H_, 1);                           // gelu(m1_ln)
    gemm(stream, x2, H_, in(87), in(86), x1, H_, nullptr, 0, 1.0f, 0, B_, H_, H_);  // m2
    ln(stream, x1, H_, in(89), in(88), x2, H_, B_, H_, 1);                           // gelu(m2_ln) -> m
    float* th  = big;
    float* thn = big + (size_t)B_ * 256;
    for (int i = 0; i < 10; ++i) {
        const int p0 = 22 + 6 * i; // h1.b h1.w h2.b h2.w h_ln.b h_ln.g
        gemm(stream, x2, H_, in(p0 + 1), in(p0 + 0), th, 256, nullptr, 0, 1.0f, 0, B_, 256, H_);
        ln(stream, th, 256, in(p0 + 5), in(p0 + 4), thn, 256, B_, 256, 1);
        gemm(stream, thn, 256, in(p0 + 3), in(p0 + 2), o_xp + (size_t)i * 500, G_,
             nullptr, 0, 1.0f, 0, B_, 500, 256);
    }
    // x_pred += 0.1 * (z @ glob_w + glob_b)   (in-place residual)
    gemm(stream, zb, L_, in(21), in(20), o_xp, G_, o_xp, G_, 0.1f, 0, B_, G_, L_);
}
